// BasePatchOrthogonalMix_6133213298738
// MI455X (gfx1250) — matrix operand order repaired and checked by reading, not compiled, not run on hardware
//
#include <hip/hip_runtime.h>
#include <hip/hip_bf16.h>

typedef __attribute__((ext_vector_type(16))) __bf16 v16bf;
typedef __attribute__((ext_vector_type(8)))  __bf16 v8bf;
typedef __attribute__((ext_vector_type(8)))  float  v8f;

#define BM 128
#define BN 128
#define BK 32
#define KTOT 1024
#define KSTEPS (KTOT / BK)     // 32
#define LDS_STRIDE 40          // 32 bf16 + 8 pad -> 80-byte row stride (16B aligned)

// x: [16, 64, 256, 256] fp32 ; W: [1024, 1024] fp32 ; y: [16, 64, 256, 256] fp32
// A[m, d] = x[b][c][hp*4+ph][wp*4+pw], m = b*4096 + hp*64 + wp, d = c*16 + ph*4 + pw
// mixed[m, e] = sum_d A[m,d] * W[e,d];  y folds mixed back with e = c*16 + ph*4 + pw.
__global__ __launch_bounds__(256, 1)
void patchmix_wmma_bf16x3(const float* __restrict__ x,
                          const float* __restrict__ Wm,
                          float* __restrict__ y) {
    __shared__ __bf16 Ah[BM * LDS_STRIDE];
    __shared__ __bf16 Al[BM * LDS_STRIDE];
    __shared__ __bf16 Bh[BN * LDS_STRIDE];
    __shared__ __bf16 Bl[BN * LDS_STRIDE];

    const int tid = threadIdx.x;
    const int bn  = blockIdx.x;   // 0..7   (N tiles of 128)
    const int bm  = blockIdx.y;   // 0..511 (M tiles of 128)

    // ---- cooperative loader mapping: 256 threads, each owns 1 row-half (16 floats) of A and B tiles
    const int lrow  = tid >> 1;   // 0..127
    const int lhalf = tid & 1;    // which 16 of the 32 k-values

    // A row -> patch coordinates
    const int m  = bm * BM + lrow;
    const int pb = m >> 12;            // / 4096
    const int hp = (m >> 6) & 63;
    const int wp = m & 63;
    const long xbase = (long)pb * 64L * 65536L + (long)(hp * 4) * 256L + (long)(wp * 4);

    // B row -> W row e
    const int e = bn * BN + lrow;
    const float* wrow = Wm + (long)e * KTOT;

    // ---- compute mapping: 8 waves, 4 in M (32 rows each) x 2 in N (64 cols each)
    const int wv   = tid >> 5;
    const int lane = tid & 31;
    const int wm_  = wv & 3;      // M wave
    const int wn_  = wv >> 2;     // N wave
    const int lr   = lane & 15;
    const int g    = lane >> 4;

    v8f acc[2][4];
    #pragma unroll
    for (int i = 0; i < 2; ++i)
        #pragma unroll
        for (int j = 0; j < 4; ++j)
            acc[i][j] = {};

    for (int kt = 0; kt < KSTEPS; ++kt) {
        const int k0 = kt * BK;

        // ---- global loads into registers (float4-aligned for both A gather and W rows)
        float4 av[4], bv[4];
        #pragma unroll
        for (int i = 0; i < 4; ++i) {
            const int dl = lhalf * 16 + i * 4;   // 0..28 step 4
            const int d  = k0 + dl;
            const int c  = d >> 4;               // channel
            const int ph = (d >> 2) & 3;         // patch row
            av[i] = *(const float4*)(x + xbase + (long)c * 65536L + (long)ph * 256L);
            bv[i] = *(const float4*)(wrow + d);
        }
        if (kt + 1 < KSTEPS) {  // prefetch next k-step (emits global_prefetch_b8)
            const int dn = k0 + BK + lhalf * 16;
            __builtin_prefetch((const void*)(wrow + dn), 0, 3);
            __builtin_prefetch((const void*)(x + xbase + (long)(dn >> 4) * 65536L
                                               + (long)((dn >> 2) & 3) * 256L), 0, 3);
        }

        __syncthreads();  // previous iteration finished reading LDS

        // ---- fp32 -> bf16 hi/lo split, stage to LDS
        #pragma unroll
        for (int i = 0; i < 4; ++i) {
            const int dl = lhalf * 16 + i * 4;
            const float* af = (const float*)&av[i];
            const float* bf = (const float*)&bv[i];
            #pragma unroll
            for (int j = 0; j < 4; ++j) {
                const float fa = af[j];
                const __bf16 fah = (__bf16)fa;
                const __bf16 fal = (__bf16)(fa - (float)fah);
                Ah[lrow * LDS_STRIDE + dl + j] = fah;
                Al[lrow * LDS_STRIDE + dl + j] = fal;
                const float fb = bf[j];
                const __bf16 fbh = (__bf16)fb;
                const __bf16 fbl = (__bf16)(fb - (float)fbh);
                Bh[lrow * LDS_STRIDE + dl + j] = fbh;
                Bl[lrow * LDS_STRIDE + dl + j] = fbl;
            }
        }
        __syncthreads();

        // ---- build fragments per ISA VGPR layouts
        // A 16x32 (bf16): lane group g holds K = g*8..g*8+7 (elems 0..7), 16+g*8..+7 (elems 8..15)
        v16bf ahf[2], alf[2];
        #pragma unroll
        for (int f = 0; f < 2; ++f) {
            const int row = wm_ * 32 + f * 16 + lr;
            const v8bf a0 = *(const v8bf*)&Ah[row * LDS_STRIDE + g * 8];
            const v8bf a1 = *(const v8bf*)&Ah[row * LDS_STRIDE + 16 + g * 8];
            const v8bf l0 = *(const v8bf*)&Al[row * LDS_STRIDE + g * 8];
            const v8bf l1 = *(const v8bf*)&Al[row * LDS_STRIDE + 16 + g * 8];
            #pragma unroll
            for (int q = 0; q < 8; ++q) {
                ahf[f][q] = a0[q]; ahf[f][q + 8] = a1[q];
                alf[f][q] = l0[q]; alf[f][q + 8] = l1[q];
            }
        }
        // B 32x16 (bf16): lane = column n (= lr), lane group g holds K = g*8..g*8+7 (elems 0..7), 16+g*8..+7 (elems 8..15)
        v16bf bhf[4], blf[4];
        #pragma unroll
        for (int f = 0; f < 4; ++f) {
            const int row = wn_ * 64 + f * 16 + lr;   // e_local
            const v8bf b0 = *(const v8bf*)&Bh[row * LDS_STRIDE + g * 8];
            const v8bf b1 = *(const v8bf*)&Bh[row * LDS_STRIDE + 16 + g * 8];
            const v8bf l0 = *(const v8bf*)&Bl[row * LDS_STRIDE + g * 8];
            const v8bf l1 = *(const v8bf*)&Bl[row * LDS_STRIDE + 16 + g * 8];
            #pragma unroll
            for (int q = 0; q < 8; ++q) {
                bhf[f][q] = b0[q]; bhf[f][q + 8] = b1[q];
                blf[f][q] = l0[q]; blf[f][q + 8] = l1[q];
            }
        }

        // ---- bf16x3 WMMA: hi*hi + lo*hi + hi*lo, fp32 accumulate
        #pragma unroll
        for (int i = 0; i < 2; ++i) {
            #pragma unroll
            for (int j = 0; j < 4; ++j) {
                acc[i][j] = __builtin_amdgcn_wmma_f32_16x16x32_bf16(
                    false, ahf[i], false, bhf[j], (short)0, acc[i][j], false, false);
                acc[i][j] = __builtin_amdgcn_wmma_f32_16x16x32_bf16(
                    false, alf[i], false, bhf[j], (short)0, acc[i][j], false, false);
                acc[i][j] = __builtin_amdgcn_wmma_f32_16x16x32_bf16(
                    false, ahf[i], false, blf[j], (short)0, acc[i][j], false, false);
            }
        }
    }

    // ---- epilogue: C/D layout is (M = v + 8*g, N = lr); fold back to y[b][c][h][w]
    #pragma unroll
    for (int i = 0; i < 2; ++i) {
        const int mbase = bm * BM + wm_ * 32 + i * 16 + g * 8;
        #pragma unroll
        for (int j = 0; j < 4; ++j) {
            const int ecol = bn * BN + wn_ * 64 + j * 16 + lr;
            const int cc = ecol >> 4;
            const int ph = (ecol >> 2) & 3;
            const int pw = ecol & 3;
            #pragma unroll
            for (int v = 0; v < 8; ++v) {
                const int mm  = mbase + v;
                const int pb2 = mm >> 12;
                const int hp2 = (mm >> 6) & 63;
                const int wp2 = mm & 63;
                const long oidx = ((long)(pb2 * 64 + cc) * 256L + (long)(hp2 * 4 + ph)) * 256L
                                  + (long)(wp2 * 4 + pw);
                y[oidx] = acc[i][j][v];
            }
        }
    }
}

extern "C" void kernel_launch(void* const* d_in, const int* in_sizes, int n_in,
                              void* d_out, int out_size, void* d_ws, size_t ws_size,
                              hipStream_t stream) {
    const float* x  = (const float*)d_in[0];   // [16,64,256,256] fp32
    const float* Wm = (const float*)d_in[1];   // [1024,1024] fp32
    float* y = (float*)d_out;                  // [16,64,256,256] fp32
    dim3 grid(8, 512);                         // (N tiles, M tiles) — N-tiles adjacent for L2 reuse of A
    dim3 block(256);
    patchmix_wmma_bf16x3<<<grid, block, 0, stream>>>(x, Wm, y);
}
